// VC_Aggregator_85048942395937
// MI455X (gfx1250) — matrix operand order repaired and checked by reading, not compiled, not run on hardware
//
#include <hip/hip_runtime.h>
#include <hip/hip_bf16.h>

// CDNA5 / gfx1250: wave32, WMMA bf16 16x16x32, f32 accumulate.
typedef __bf16  v16bf __attribute__((ext_vector_type(16)));
typedef float   v8f   __attribute__((ext_vector_type(8)));

#define NNODES 16384
#define LHIST  50
#define DDIM   64
#define LPAD   64   // L padded to 4 M-tiles of 16

// Fragment-ordered bf16 weight workspace layout (elements):
//   W1 frags: [kc(4)][nt(4)][lane(32)][e(16)]  -> 8192 elems @ offset 0
//   Wk frags: [kc(2)][nt(4)][lane(32)][e(16)]  -> 4096 elems @ offset 8192
//   Wv frags: [kc(2)][nt(4)][lane(32)][e(16)]  -> 4096 elems @ offset 12288
#define WS_W1 0
#define WS_WK 8192
#define WS_WV 12288
#define WS_TOTAL 16384

// B-matrix 32x16 bf16 lane layout: lane = (n%16) + 16*hb,
// elements 0..7 -> K = kc*32 + hb*8 + e ; elements 8..15 -> +16 (same as A).
// A-matrix 16x32 bf16 lane layout: lane = (m%16) + 16*hb,
// elements 0..7 -> K = kc*32 + hb*8 + e ; elements 8..15 -> +16.

// ---------------------------------------------------------------------------
// One-shot per launch: convert fp32 weights to fragment-ordered bf16.
// ---------------------------------------------------------------------------
__global__ __launch_bounds__(256)
void VC_Aggregator_weights_kernel(const float* __restrict__ W1,
                                  const float* __restrict__ Wk,
                                  const float* __restrict__ Wv,
                                  __bf16* __restrict__ ws) {
    const int i = blockIdx.x * 256 + threadIdx.x;     // 0..16383
    if (i >= WS_TOTAL) return;
    const float* W;
    int j;
    if (i < WS_WK)      { W = W1; j = i;         }
    else if (i < WS_WV) { W = Wk; j = i - WS_WK; }
    else                { W = Wv; j = i - WS_WV; }
    const int e    = j & 15;
    const int lane = (j >> 4) & 31;
    const int nt   = (j >> 9) & 3;
    const int kc   = j >> 11;                         // 0..3 (W1), 0..1 (Wk/Wv)
    const int ln   = lane & 15;
    const int hb   = lane >> 4;
    ws[i] = (__bf16)W[(size_t)(kc * 32 + hb * 8 + (e & 7) + (e >> 3) * 16) * DDIM + nt * 16 + ln];
}

// ---------------------------------------------------------------------------
// One workgroup (256 threads = 8 waves) per node. ~60KB LDS -> ~5 WG/WGP.
// ---------------------------------------------------------------------------
__global__ __launch_bounds__(256)
void VC_Aggregator_85048942395937_kernel(
    const int*   __restrict__ nodes,       // [N]
    const int*   __restrict__ history_vc,  // [N,50]
    const int*   __restrict__ history_r,   // [N,50]
    const float* __restrict__ c2e,         // [1000,64]
    const float* __restrict__ v2e,         // [100000,64]
    const float* __restrict__ r2e,         // [5,64]
    const float* __restrict__ b1,          // [64]
    const float* __restrict__ Wq,          // [64,64]
    const float* __restrict__ bq,          // [64]
    const float* __restrict__ bk,          // [64]
    const float* __restrict__ bv,          // [64]
    const float* __restrict__ Wo,          // [64,64]
    const float* __restrict__ bo,          // [64]
    const __bf16* __restrict__ wsW,        // fragment-ordered bf16 weights
    float*       __restrict__ out)         // [N,64]
{
    // ---- LDS (fragment-ordered bf16 for WMMA inputs) ----
    __shared__ __align__(32) __bf16 xs_frag[4 * 4 * 32 * 16];  // 16KB: [mt][kc(4)][lane][e]
    __shared__ __align__(32) __bf16 hs_frag[4 * 2 * 32 * 16];  //  8KB: [mt][kc(2)][lane][e]
    __shared__ float  ks_s[LPAD][68];       // k = h@Wk+bk, f32 row-major
    __shared__ float  vs_s[LPAD][68];       // v = h@Wv+bv, f32 row-major
    __shared__ float  vrep_s[DDIM];
    __shared__ float  q_s[DDIM];
    __shared__ float  sc_s[LPAD];
    __shared__ float  ctx_s[DDIM];
    __shared__ int    hvc_s[LHIST];
    __shared__ int    hr_s[LHIST];
    __shared__ float  red_s[2];

    const int n    = blockIdx.x;
    const int tid  = threadIdx.x;
    const int wv   = tid >> 5;        // wave id 0..7
    const int lane = tid & 31;
    const int ln   = lane & 15;
    const int hb   = lane >> 4;

    // ---- Stage 0: indices + node embedding ----
    if (tid < LHIST) hvc_s[tid] = history_vc[n * LHIST + tid];
    if (tid >= 64 && tid < 64 + LHIST) hr_s[tid - 64] = history_r[n * LHIST + tid - 64];
    if (tid >= 128 && tid < 128 + DDIM)
        vrep_s[tid - 128] = v2e[(size_t)nodes[n] * DDIM + (tid - 128)];
    __syncthreads();

    // ---- Stage 0b: gather x = [c2e|r2e] -> LDS bf16 fragments, packed pair stores
    {
        uint* xsw = (uint*)xs_frag;
        for (int p = tid; p < LPAD * 64; p += 256) {       // 64 col-pairs per row
            const int row  = p >> 6;
            const int col0 = (p & 63) * 2;
            float f0 = 0.f, f1 = 0.f;
            if (row < LHIST) {
                const float2 t = (col0 < DDIM)
                    ? *(const float2*)&c2e[(size_t)hvc_s[row] * DDIM + col0]
                    : *(const float2*)&r2e[(size_t)hr_s[row]  * DDIM + (col0 - DDIM)];
                f0 = t.x; f1 = t.y;
            }
            const int mt  = row >> 4, l = row & 15;
            const int kc  = col0 >> 5, ko = col0 & 31;
            const int hb2 = (ko >> 3) & 1;
            const int eb  = (ko < 16) ? (ko & 7) : (8 + (ko & 7));   // even
            const int idx = (((mt * 4 + kc) * 32) + (l + hb2 * 16)) * 16 + eb;
            const unsigned short u0 = __builtin_bit_cast(unsigned short, (__bf16)f0);
            const unsigned short u1 = __builtin_bit_cast(unsigned short, (__bf16)f1);
            xsw[idx >> 1] = (uint)u0 | ((uint)u1 << 16);
        }
    }
    __syncthreads();

    // ---- Stage 1: h = relu(x @ W1 + b1)  -> hs_frag
    {
        const int mt     = wv & 3;
        const int ntBase = (wv >> 2) * 2;
        for (int nti = 0; nti < 2; ++nti) {
            const int nt   = ntBase + nti;
            const int colN = nt * 16 + ln;
            v8f acc = {};
            for (int kc = 0; kc < 4; ++kc) {
                v16bf a = *(const v16bf*)&xs_frag[(((mt * 4 + kc) * 32) + lane) * 16];
                v16bf b = *(const v16bf*)&wsW[WS_W1 + (((kc * 4 + nt) * 32) + lane) * 16];
                acc = __builtin_amdgcn_wmma_f32_16x16x32_bf16(
                          false, a, false, b, (short)0, acc, false, false);
            }
            const float bias = b1[colN];
            const int kcH = colN >> 5, koH = colN & 31;
            const int hb2 = (koH >> 3) & 1;
            const int eb  = (koH < 16) ? (koH & 7) : (8 + (koH & 7));
#pragma unroll
            for (int r = 0; r < 8; ++r) {
                const int l = r + hb * 8;                    // C/D: lanes 16-31 -> M+8
                float h = acc[r] + bias;
                hs_frag[(((mt * 2 + kcH) * 32) + (l + hb2 * 16)) * 16 + eb] =
                    (__bf16)(h > 0.f ? h : 0.f);
            }
        }
    }
    __syncthreads();

    // ---- Stage 2: k = h@Wk + bk (waves 0-3), v = h@Wv + bv (waves 4-7)
    {
        const bool isK = (wv < 4);
        const __bf16* wsM = isK ? (wsW + WS_WK) : (wsW + WS_WV);
        const float*  bvec = isK ? bk : bv;
        float (*dst)[68]   = isK ? ks_s : vs_s;
        const int mt = wv & 3;
        for (int nt = 0; nt < 4; ++nt) {
            const int colN = nt * 16 + ln;
            v8f acc = {};
            for (int kc = 0; kc < 2; ++kc) {
                v16bf a = *(const v16bf*)&hs_frag[(((mt * 2 + kc) * 32) + lane) * 16];
                v16bf b = *(const v16bf*)&wsM[(((kc * 4 + nt) * 32) + lane) * 16];
                acc = __builtin_amdgcn_wmma_f32_16x16x32_bf16(
                          false, a, false, b, (short)0, acc, false, false);
            }
            const float bias = bvec[colN];
#pragma unroll
            for (int r = 0; r < 8; ++r)
                dst[mt * 16 + r + hb * 8][colN] = acc[r] + bias;
        }
    }
    __syncthreads();

    // ---- Stage 3: q = vrep @ Wq + bq  (tiny VALU)
    if (tid < DDIM) {
        float s = bq[tid];
        for (int j = 0; j < DDIM; ++j) s += vrep_s[j] * Wq[(size_t)j * DDIM + tid];
        q_s[tid] = s;
    }
    __syncthreads();

    // ---- Stage 4: scores + softmax over L=50
    if (tid < LHIST) {
        float s = 0.f;
        for (int d = 0; d < DDIM; ++d) s += q_s[d] * ks_s[tid][d];
        sc_s[tid] = s * 0.125f;                 // 1/sqrt(64)
    }
    __syncthreads();
    if (tid == 0) {
        float m = -3.0e38f;
        for (int l = 0; l < LHIST; ++l) m = fmaxf(m, sc_s[l]);
        red_s[0] = m;
    }
    __syncthreads();
    if (tid < LHIST) sc_s[tid] = __expf(sc_s[tid] - red_s[0]);
    __syncthreads();
    if (tid == 0) {
        float s = 0.f;
        for (int l = 0; l < LHIST; ++l) s += sc_s[l];
        red_s[1] = 1.f / s;
    }
    __syncthreads();

    // ---- Stage 5: ctx = softmax(scores) @ v
    if (tid < DDIM) {
        float c = 0.f;
        for (int l = 0; l < LHIST; ++l) c += sc_s[l] * vs_s[l][tid];
        ctx_s[tid] = c * red_s[1];
    }
    __syncthreads();

    // ---- Stage 6: out = ctx @ Wo + bo
    if (tid < DDIM) {
        float o = bo[tid];
        for (int j = 0; j < DDIM; ++j) o += ctx_s[j] * Wo[(size_t)j * DDIM + tid];
        out[(size_t)n * DDIM + tid] = o;
    }
}

extern "C" void kernel_launch(void* const* d_in, const int* in_sizes, int n_in,
                              void* d_out, int out_size, void* d_ws, size_t ws_size,
                              hipStream_t stream) {
    (void)n_in; (void)out_size; (void)ws_size;
    const int*   nodes      = (const int*)  d_in[0];
    const int*   history_vc = (const int*)  d_in[1];
    const int*   history_r  = (const int*)  d_in[2];
    const float* c2e        = (const float*)d_in[3];
    const float* v2e        = (const float*)d_in[4];
    const float* r2e        = (const float*)d_in[5];
    const float* W1         = (const float*)d_in[6];
    const float* b1         = (const float*)d_in[7];
    const float* Wq         = (const float*)d_in[8];
    const float* bq         = (const float*)d_in[9];
    const float* Wk         = (const float*)d_in[10];
    const float* bk         = (const float*)d_in[11];
    const float* Wv         = (const float*)d_in[12];
    const float* bv         = (const float*)d_in[13];
    const float* Wo         = (const float*)d_in[14];
    const float* bo         = (const float*)d_in[15];
    float*  out = (float*)d_out;
    __bf16* ws  = (__bf16*)d_ws;            // 32KB used

    // Pass 1: fragment-ordered bf16 weights (shared by all nodes).
    VC_Aggregator_weights_kernel<<<(WS_TOTAL + 255) / 256, 256, 0, stream>>>(W1, Wk, Wv, ws);

    // Pass 2: one workgroup per node.
    const int N = in_sizes[0];
    VC_Aggregator_85048942395937_kernel<<<N, 256, 0, stream>>>(
        nodes, history_vc, history_r, c2e, v2e, r2e,
        b1, Wq, bq, bk, bv, Wo, bo, ws, out);
}
